// LocalNeighborGATDiscreteSAC_25494925869603
// MI455X (gfx1250) — matrix operand order repaired and checked by reading, not compiled, not run on hardware
//
#include <hip/hip_runtime.h>
#include <hip/hip_bf16.h>
#include <math.h>

typedef __attribute__((ext_vector_type(16))) _Float16 v16h;
typedef __attribute__((ext_vector_type(8)))  float    v8f;

// ---------------------------------------------------------------------------
// Monotone float<->uint key for atomicMax-based segment max
// ---------------------------------------------------------------------------
__device__ __forceinline__ unsigned f2key(float f) {
    unsigned b = __float_as_uint(f);
    return b ^ ((((int)b) >> 31) | 0x80000000u);
}
__device__ __forceinline__ float key2f(unsigned k) {
    unsigned b = (k & 0x80000000u) ? (k ^ 0x80000000u) : ~k;
    return __uint_as_float(b);
}

// fast ELU: arg of exp is <= 0, __expf (v_exp_f32) is plenty accurate there
__device__ __forceinline__ float elu1(float v) { return v > 0.f ? v : (__expf(v) - 1.f); }

template <int ACT>
__device__ __forceinline__ float actf(float v) {
    if (ACT == 1) return v > 0.f ? v : 0.f;
    if (ACT == 2) return elu1(v);
    return v;
}

// ---------------------------------------------------------------------------
// WMMA GEMM:  C[rows x NC] = act(A[rows x K](f32->f16) @ W[K x NC](f32->f16) + bias)
// One wave per 16-row tile; 8 waves/block. Weights are staged once per block
// into LDS, converted to f16 and pre-swizzled into B-fragment lane order so
// each wave reads a fragment as one contiguous 32-byte DS load.
// ACT: 0=none 1=relu 2=elu (compile-time -> no dispatch branches in epilogue)
// ---------------------------------------------------------------------------
template <int K, int ACT>
__global__ void __launch_bounds__(256)
gemm_wmma_f16(const float* __restrict__ A, int lda,
              const float* __restrict__ W, int NC,
              const float* __restrict__ bias,
              float* __restrict__ C, int ldc, int ccol0,
              int nrows)
{
    constexpr int NKT   = K / 32;
    constexpr int MAXNT = 6;                      // up to NC = 96
    __shared__ v16h sB[NKT * MAXNT * 32];         // fragment (kt,nt) x 32 lanes x 16 halves

    const int ntiles = NC >> 4;

    // ---- cooperative stage: W[k][n] -> sB[(kt*ntiles+nt)*32 + half*16 + (n&15)][j]
    //      where k = kt*32 + half*8 + j (j<8), kt*32 + 16 + half*8 + (j-8) (j>=8)
    {
        _Float16* sBh = (_Float16*)sB;
        const int total = K * NC;
        for (int i = (int)threadIdx.x; i < total; i += (int)blockDim.x) {
            const int k  = i / NC;
            const int n  = i - k * NC;
            const int kt   = k >> 5;
            const int half = (k >> 3) & 1;
            const int j    = (k & 7) + (((k >> 4) & 1) << 3);
            const int nt   = n >> 4;
            const int ln   = n & 15;
            sBh[((((kt * ntiles + nt) << 5) + (half << 4) + ln) << 4) + j] = (_Float16)W[i];
        }
    }
    __syncthreads();

    const int wave = (int)((blockIdx.x * blockDim.x + threadIdx.x) >> 5);
    const int lane = (int)(threadIdx.x & 31);
    const int m0 = wave * 16;
    if (m0 >= nrows) return;                 // wave-uniform exit (after barrier)
    const int half = lane >> 4;              // 0: lanes 0-15, 1: lanes 16-31
    const int lo   = lane & 15;
    const bool fullTile = (m0 + 16 <= nrows);   // wave-uniform

    int m = m0 + lo;                         // A-fragment row for this lane
    if (m >= nrows) m = nrows - 1;           // clamp (garbage rows never stored)

    // ---- load & convert A fragments (16-bit A 16x32 layout, ISA 7.12.2) ----
    v16h afrag[NKT];
    const float* arow = A + (size_t)m * (size_t)lda;
#pragma unroll
    for (int kt = 0; kt < NKT; ++kt) {
        const int kb = kt * 32 + half * 8;
        float4 p0 = *(const float4*)(arow + kb);
        float4 p1 = *(const float4*)(arow + kb + 4);
        float4 p2 = *(const float4*)(arow + kb + 16);
        float4 p3 = *(const float4*)(arow + kb + 20);
        v16h a;
        a[0]=(_Float16)p0.x; a[1]=(_Float16)p0.y; a[2]=(_Float16)p0.z; a[3]=(_Float16)p0.w;
        a[4]=(_Float16)p1.x; a[5]=(_Float16)p1.y; a[6]=(_Float16)p1.z; a[7]=(_Float16)p1.w;
        a[8]=(_Float16)p2.x; a[9]=(_Float16)p2.y; a[10]=(_Float16)p2.z; a[11]=(_Float16)p2.w;
        a[12]=(_Float16)p3.x; a[13]=(_Float16)p3.y; a[14]=(_Float16)p3.z; a[15]=(_Float16)p3.w;
        afrag[kt] = a;
    }

    for (int nt = 0; nt < ntiles; ++nt) {
        const int n = nt * 16 + lo;          // output column for this lane
        const float bv = bias[n];            // bias depends on N only
        v8f acc;
#pragma unroll
        for (int r = 0; r < 8; ++r) acc[r] = bv;

#pragma unroll
        for (int kt = 0; kt < NKT; ++kt) {
            const v16h b = sB[(kt * ntiles + nt) * 32 + lane];   // one 32B DS read
            acc = __builtin_amdgcn_wmma_f32_16x16x32_f16(
                false, afrag[kt], false, b, (short)0, acc, false, false);
        }

        // C/D layout: element r -> (M = m0 + 8*half + r, N = n)
        float* cbase = C + (size_t)(m0 + half * 8) * (size_t)ldc + ccol0 + n;
        if (fullTile) {                      // common path: no per-element guards
#pragma unroll
            for (int r = 0; r < 8; ++r)
                cbase[(size_t)r * (size_t)ldc] = actf<ACT>(acc[r]);
        } else {                             // last partial tile only
#pragma unroll
            for (int r = 0; r < 8; ++r) {
                if (m0 + half * 8 + r < nrows)
                    cbase[(size_t)r * (size_t)ldc] = actf<ACT>(acc[r]);
            }
        }
    }
}

// ---------------------------------------------------------------------------
// Self-loop edge_attr = mean of incoming edge_attr (PyG fill_value='mean')
// ---------------------------------------------------------------------------
__global__ void ea_mean_accum(const int* __restrict__ ei, const float* __restrict__ ea,
                              float* __restrict__ cnt, float* __restrict__ easum, int E)
{
    int e = (int)(blockIdx.x * blockDim.x + threadIdx.x);
    if (e >= E) return;
    int d = ei[E + e];                        // edge_index[1][e]
    atomicAdd(cnt + d, 1.f);
    atomicAdd(easum + 2 * d + 0, ea[2 * e + 0]);
    atomicAdd(easum + 2 * d + 1, ea[2 * e + 1]);
}

__global__ void ea_mean_fin(const float* __restrict__ cnt, const float* __restrict__ easum,
                            float* __restrict__ eam, int N)
{
    int i = (int)(blockIdx.x * blockDim.x + threadIdx.x);
    if (i >= N) return;
    float c = cnt[i]; c = c > 1.f ? c : 1.f;
    eam[2 * i + 0] = easum[2 * i + 0] / c;
    eam[2 * i + 1] = easum[2 * i + 1] / c;
}

// ---------------------------------------------------------------------------
// Edge pass 1: logits per (edge, head) + segment max (atomicMax on keys)
// ---------------------------------------------------------------------------
__global__ void gat_logits(const int* __restrict__ ei, const float* __restrict__ ea,
                           const float* __restrict__ eam,
                           const float* __restrict__ xl, const float* __restrict__ xr,
                           const float* __restrict__ we, const float* __restrict__ att,
                           float* __restrict__ lbuf, unsigned* __restrict__ mxkey,
                           int E, int N)
{
    __shared__ float s_we[128];   // we: [2][64]
    __shared__ float s_att[64];   // att: [2][32]
    int t = (int)threadIdx.x;
    if (t < 128) s_we[t] = we[t];
    if (t < 64)  s_att[t] = att[t];
    __syncthreads();

    int e  = (int)(blockIdx.x * blockDim.x + t);
    int E2 = E + N;
    if (e >= E2) return;

    int s, d; float a0, a1;
    if (e < E) { s = ei[e]; d = ei[E + e]; a0 = ea[2 * e]; a1 = ea[2 * e + 1]; }
    else       { s = e - E; d = s;         a0 = eam[2 * s]; a1 = eam[2 * s + 1]; }

    const float* xls = xl + (size_t)s * 64;
    const float* xrd = xr + (size_t)d * 64;
#pragma unroll
    for (int h = 0; h < 2; ++h) {
        float acc = 0.f;
#pragma unroll
        for (int c = 0; c < 32; ++c) {
            int ch = h * 32 + c;
            float mv = xls[ch] + xrd[ch] + a0 * s_we[ch] + a1 * s_we[64 + ch];
            float sv = mv > 0.f ? mv : 0.2f * mv;        // leaky_relu(0.2)
            acc += sv * s_att[ch];
        }
        lbuf[2 * e + h] = acc;
        atomicMax(mxkey + 2 * d + h, f2key(acc));
    }
}

// ---------------------------------------------------------------------------
// Edge pass 2: ex = exp(logit - mx[dst]); den[dst] += ex (lbuf overwritten)
// ---------------------------------------------------------------------------
__global__ void gat_exp(const int* __restrict__ ei, const unsigned* __restrict__ mxkey,
                        float* __restrict__ lbuf, float* __restrict__ den, int E, int N)
{
    int e = (int)(blockIdx.x * blockDim.x + threadIdx.x);
    if (e >= E + N) return;
    int d = (e < E) ? ei[E + e] : (e - E);
#pragma unroll
    for (int h = 0; h < 2; ++h) {
        float ex = __expf(lbuf[2 * e + h] - key2f(mxkey[2 * d + h]));  // arg <= 0
        lbuf[2 * e + h] = ex;
        atomicAdd(den + 2 * d + h, ex);
    }
}

// ---------------------------------------------------------------------------
// Edge pass 3: agg[dst] += alpha * xl[src]  (per-channel f32 atomics)
// ---------------------------------------------------------------------------
__global__ void gat_agg(const int* __restrict__ ei, const float* __restrict__ lbuf,
                        const float* __restrict__ den, const float* __restrict__ xl,
                        float* __restrict__ agg, int E, int N)
{
    int e = (int)(blockIdx.x * blockDim.x + threadIdx.x);
    if (e >= E + N) return;
    int s, d;
    if (e < E) { s = ei[e]; d = ei[E + e]; } else { s = e - E; d = s; }
    float al0 = lbuf[2 * e + 0] / den[2 * d + 0];
    float al1 = lbuf[2 * e + 1] / den[2 * d + 1];
    const float* xls = xl + (size_t)s * 64;
    float* ad = agg + (size_t)d * 64;
#pragma unroll
    for (int c = 0; c < 32; ++c) atomicAdd(ad + c,      al0 * xls[c]);
#pragma unroll
    for (int c = 0; c < 32; ++c) atomicAdd(ad + 32 + c, al1 * xls[32 + c]);
}

// ---------------------------------------------------------------------------
// h2 = elu(agg + gat_b)
// ---------------------------------------------------------------------------
__global__ void gat_h2(const float* __restrict__ agg, const float* __restrict__ gat_b,
                       float* __restrict__ h2, int total)
{
    int i = (int)(blockIdx.x * blockDim.x + threadIdx.x);
    if (i >= total) return;
    float v = agg[i] + gat_b[i & 63];
    h2[i] = elu1(v);
}

// ---------------------------------------------------------------------------
extern "C" void kernel_launch(void* const* d_in, const int* in_sizes, int n_in,
                              void* d_out, int out_size, void* d_ws, size_t ws_size,
                              hipStream_t stream)
{
    const float* obs   = (const float*)d_in[0];
    const int*   ei    = (const int*)  d_in[1];   // [2,E] (jax int64 -> int32 w/o x64)
    const float* ea    = (const float*)d_in[2];   // [E,2]
    const float* le_w1 = (const float*)d_in[3];
    const float* le_b1 = (const float*)d_in[4];
    const float* le_w2 = (const float*)d_in[5];
    const float* le_b2 = (const float*)d_in[6];
    const float* ip_w  = (const float*)d_in[7];
    const float* ip_b  = (const float*)d_in[8];
    const float* wl    = (const float*)d_in[9];
    const float* bl    = (const float*)d_in[10];
    const float* wr    = (const float*)d_in[11];
    const float* br    = (const float*)d_in[12];
    const float* we    = (const float*)d_in[13];
    const float* att   = (const float*)d_in[14];
    const float* gat_b = (const float*)d_in[15];
    const float* op_w  = (const float*)d_in[16];
    const float* op_b  = (const float*)d_in[17];
    const float* f_w1  = (const float*)d_in[18];
    const float* f_b1  = (const float*)d_in[19];
    const float* f_w2  = (const float*)d_in[20];
    const float* f_b2  = (const float*)d_in[21];
    float* out = (float*)d_out;

    const int N  = in_sizes[0] / 128;   // OBS = 128
    const int E  = in_sizes[2] / 2;     // EDIM = 2
    const int E2 = E + N;

    // ---- workspace carve-out (256B aligned chunks) ----
    size_t off = 0;
    char* base = (char*)d_ws;
    auto carve = [&](size_t bytes) -> void* {
        void* p = base + off;
        off += (bytes + 255) & ~(size_t)255;
        return p;
    };
    float*    t1    = (float*)   carve((size_t)N * 64 * 4);  // relu(obs@le_w1); later reused as h2
    float*    h     = (float*)   carve((size_t)N * 64 * 4);
    float*    xl    = (float*)   carve((size_t)N * 64 * 4);
    float*    xr    = (float*)   carve((size_t)N * 64 * 4);
    float*    agg   = (float*)   carve((size_t)N * 64 * 4);
    float*    comb  = (float*)   carve((size_t)N * 96 * 4);
    float*    t2    = (float*)   carve((size_t)N * 96 * 4);
    float*    stats = (float*)   carve((size_t)N * 7 * 4);   // cnt | easum | mxkey | den (contig)
    float*    eam   = (float*)   carve((size_t)N * 2 * 4);
    float*    lbuf  = (float*)   carve((size_t)E2 * 2 * 4);
    float*    cnt   = stats;
    float*    easum = stats + N;
    unsigned* mxkey = (unsigned*)(stats + 3 * (size_t)N);
    float*    den   = stats + 5 * (size_t)N;
    float*    h2    = t1;   // t1 consumed before h2 is written (stream-ordered)
    (void)ws_size; (void)n_in; (void)out_size;

    // ---- zero accumulators (graph-capture-safe memset nodes) ----
    hipMemsetAsync(agg,   0, (size_t)N * 64 * 4, stream);
    hipMemsetAsync(stats, 0, (size_t)N * 7 * 4,  stream);   // mxkey=0 < any valid key

    const int BT = 256;
    const int gemmBlocks = ((N + 15) / 16 + 7) / 8;         // 8 waves (16-row tiles) per block
    const int nodeBlocks = (N + BT - 1) / BT;
    const int edgeBlocks = (E + BT - 1) / BT;
    const int e2Blocks   = (E2 + BT - 1) / BT;

    // ---- LocalEncoder ----
    gemm_wmma_f16<128,1><<<gemmBlocks, BT, 0, stream>>>(obs, 128, le_w1, 64, le_b1, t1, 64, 0, N);
    gemm_wmma_f16<64,2> <<<gemmBlocks, BT, 0, stream>>>(t1,   64, le_w2, 48, le_b2, comb, 96, 0, N);

    // ---- NeighborEncoder input proj + GAT linear parts ----
    gemm_wmma_f16<128,2><<<gemmBlocks, BT, 0, stream>>>(obs, 128, ip_w, 64, ip_b, h,  64, 0, N);
    gemm_wmma_f16<64,0> <<<gemmBlocks, BT, 0, stream>>>(h,    64, wl,   64, bl,   xl, 64, 0, N);
    gemm_wmma_f16<64,0> <<<gemmBlocks, BT, 0, stream>>>(h,    64, wr,   64, br,   xr, 64, 0, N);

    // ---- self-loop edge_attr mean ----
    ea_mean_accum<<<edgeBlocks, BT, 0, stream>>>(ei, ea, cnt, easum, E);
    ea_mean_fin  <<<nodeBlocks, BT, 0, stream>>>(cnt, easum, eam, N);

    // ---- GATv2 attention: logits/max -> exp/sum -> weighted aggregate ----
    gat_logits<<<e2Blocks, BT, 0, stream>>>(ei, ea, eam, xl, xr, we, att, lbuf, mxkey, E, N);
    gat_exp   <<<e2Blocks, BT, 0, stream>>>(ei, mxkey, lbuf, den, E, N);
    gat_agg   <<<e2Blocks, BT, 0, stream>>>(ei, lbuf, den, xl, agg, E, N);

    // ---- h2 = elu(agg + gat_b); z_neighbor = h2 @ op_w + op_b -> comb[:,48:96] ----
    gat_h2<<<((N * 64) + BT - 1) / BT, BT, 0, stream>>>(agg, gat_b, h2, N * 64);
    gemm_wmma_f16<64,0><<<gemmBlocks, BT, 0, stream>>>(h2, 64, op_w, 48, op_b, comb, 96, 48, N);

    // ---- FusionMLP ----
    gemm_wmma_f16<96,1><<<gemmBlocks, BT, 0, stream>>>(comb, 96, f_w1, 96, f_b1, t2,  96, 0, N);
    gemm_wmma_f16<96,2><<<gemmBlocks, BT, 0, stream>>>(t2,   96, f_w2, 64, f_b2, out, 64, 0, N);
}
